// SelectiveSSMCell_73134703116522
// MI455X (gfx1250) — matrix operand order repaired and checked by reading, not compiled, not run on hardware
//
#include <hip/hip_runtime.h>

#define D_MODEL 2048
#define D_STATE 16
#define BATCH   2048

typedef __attribute__((ext_vector_type(16))) __bf16 v16bf;
typedef __attribute__((ext_vector_type(8)))  __bf16 bf8;
typedef __attribute__((ext_vector_type(4)))  __bf16 bf4;
typedef __attribute__((ext_vector_type(8)))  float  v8f;
typedef __attribute__((ext_vector_type(4)))  float  f4;

// ---------------------------------------------------------------------------
// Kernel 0: A_neg = -exp(A_log)   (D_MODEL x D_STATE, tiny)
// ---------------------------------------------------------------------------
__global__ __launch_bounds__(256)
void ssm_prep_A_kernel(const float* __restrict__ A_log, float* __restrict__ Aneg, int n)
{
    int i = blockIdx.x * 256 + threadIdx.x;
    if (i < n) Aneg[i] = -__expf(A_log[i]);
}

// ---------------------------------------------------------------------------
// Kernel 1: B_t = x @ W_B^T, C_t = x @ W_C^T   (BATCH x 16 each, FLOP-trivial)
// One wave (32 lanes) per batch row: lanes 0..15 -> W_B rows, 16..31 -> W_C.
// ---------------------------------------------------------------------------
__global__ __launch_bounds__(256)
void ssm_bc_kernel(const float* __restrict__ X,
                   const float* __restrict__ WB,
                   const float* __restrict__ WC,
                   float* __restrict__ Bt,
                   float* __restrict__ Ct)
{
    const int tid = threadIdx.x;
    const int row = blockIdx.x * 8 + (tid >> 5);
    const int s   = tid & 31;
    const float* w = ((s < 16) ? WB : WC) + (size_t)(s & 15) * D_MODEL;
    const float* x = X + (size_t)row * D_MODEL;

    f4 acc = {0.f, 0.f, 0.f, 0.f};
#pragma unroll 8
    for (int k = 0; k < D_MODEL; k += 4) {
        f4 xv = *(const f4*)(x + k);
        f4 wv = *(const f4*)(w + k);
        acc.x += xv.x * wv.x;
        acc.y += xv.y * wv.y;
        acc.z += xv.z * wv.z;
        acc.w += xv.w * wv.w;
    }
    float r = (acc.x + acc.y) + (acc.z + acc.w);
    if (s < 16) Bt[row * D_STATE + s]        = r;
    else        Ct[row * D_STATE + (s - 16)] = r;
}

// ---------------------------------------------------------------------------
// Kernel 2: Delta = softplus(x @ W_Delta^T + b)
// BF16-split (hi/lo) WMMA GEMM: acc += hi*hi + hi*lo + lo*hi  (near-f32 precision)
// f32 -> bf16 split happens ONCE at LDS-store time; LDS holds bf16 hi/lo tiles.
// Tile: 128x128 per workgroup (8 waves, each 32 rows x 64 cols), K-step 64.
// ---------------------------------------------------------------------------
#define TILE_M 128
#define TILE_N 128
#define TILE_K 64
#define LDSTRH 72   // bf16 row stride: 64 + 8 pad = 144 B, keeps 16B alignment

// Split one float4 into bf16 hi / lo quads.
#define SPLIT_F4(F, H, L)                                   \
    do {                                                    \
        __bf16 _h0 = (__bf16)(F).x, _h1 = (__bf16)(F).y;    \
        __bf16 _h2 = (__bf16)(F).z, _h3 = (__bf16)(F).w;    \
        (H)[0] = _h0; (H)[1] = _h1; (H)[2] = _h2; (H)[3] = _h3;          \
        (L)[0] = (__bf16)((F).x - (float)_h0);              \
        (L)[1] = (__bf16)((F).y - (float)_h1);              \
        (L)[2] = (__bf16)((F).z - (float)_h2);              \
        (L)[3] = (__bf16)((F).w - (float)_h3);              \
    } while (0)

#define CAT16(R0, R1) __builtin_shufflevector((R0), (R1), 0,1,2,3,4,5,6,7,8,9,10,11,12,13,14,15)

__global__ __launch_bounds__(256)
void ssm_delta_gemm_softplus_kernel(const float* __restrict__ X,    // (BATCH, D_MODEL)
                                    const float* __restrict__ W,    // (D_MODEL, D_MODEL) row n holds k-contig
                                    const float* __restrict__ bias, // (D_MODEL)
                                    float* __restrict__ Delta)      // (BATCH, D_MODEL)
{
    __shared__ __bf16 ldsXhi[TILE_M * LDSTRH];
    __shared__ __bf16 ldsXlo[TILE_M * LDSTRH];
    __shared__ __bf16 ldsWhi[TILE_N * LDSTRH];
    __shared__ __bf16 ldsWlo[TILE_N * LDSTRH];

    const int tid  = threadIdx.x;
    const int lane = tid & 31;
    const int wid  = tid >> 5;     // 0..7
    const int wm   = wid & 3;      // 4 wave-rows  * 32 rows = 128
    const int wn   = wid >> 2;     // 2 wave-cols  * 64 cols = 128
    const int l16  = lane & 15;
    const int half = lane >> 4;

    const int bm = blockIdx.x * TILE_M;
    const int bn = blockIdx.y * TILE_N;

    v8f acc[2][4];
    const v8f zacc = {0.f, 0.f, 0.f, 0.f, 0.f, 0.f, 0.f, 0.f};
#pragma unroll
    for (int mt = 0; mt < 2; ++mt)
#pragma unroll
        for (int nt = 0; nt < 4; ++nt) acc[mt][nt] = zacc;

    for (int kt = 0; kt < D_MODEL; kt += TILE_K) {
        // Cooperative load: 128x64 f32 per matrix, split to bf16 hi/lo at store.
        // 2048 float4s per matrix / 256 threads = 8 each.
#pragma unroll
        for (int i = 0; i < 8; ++i) {
            const int fid = tid + i * 256;        // 0..2047
            const int row = fid >> 4;             // 0..127
            const int kq  = (fid & 15) * 4;       // 0,4,..,60
            const f4 xv = *(const f4*)(X + (size_t)(bm + row) * D_MODEL + kt + kq);
            const f4 wv = *(const f4*)(W + (size_t)(bn + row) * D_MODEL + kt + kq);
            bf4 xh, xl, wh, wl;
            SPLIT_F4(xv, xh, xl);
            SPLIT_F4(wv, wh, wl);
            *(bf4*)(ldsXhi + row * LDSTRH + kq) = xh;
            *(bf4*)(ldsXlo + row * LDSTRH + kq) = xl;
            *(bf4*)(ldsWhi + row * LDSTRH + kq) = wh;
            *(bf4*)(ldsWlo + row * LDSTRH + kq) = wl;
            if (kt + TILE_K < D_MODEL) {
                __builtin_prefetch(X + (size_t)(bm + row) * D_MODEL + kt + TILE_K + kq, 0, 0);
                __builtin_prefetch(W + (size_t)(bn + row) * D_MODEL + kt + TILE_K + kq, 0, 0);
            }
        }
        __syncthreads();

#pragma unroll
        for (int ks = 0; ks < TILE_K; ks += 32) {
            // A fragments (16x32 bf16): lane l16 = M; k runs [ks+half*8,+8) and [ks+16+half*8,+8)
            v16bf a_hi[2], a_lo[2];
#pragma unroll
            for (int mt = 0; mt < 2; ++mt) {
                const int ro = (wm * 32 + mt * 16 + l16) * LDSTRH;
                const bf8 h0 = *(const bf8*)(ldsXhi + ro + ks + half * 8);
                const bf8 h1 = *(const bf8*)(ldsXhi + ro + ks + 16 + half * 8);
                const bf8 l0 = *(const bf8*)(ldsXlo + ro + ks + half * 8);
                const bf8 l1 = *(const bf8*)(ldsXlo + ro + ks + 16 + half * 8);
                a_hi[mt] = CAT16(h0, h1);
                a_lo[mt] = CAT16(l0, l1);
            }
            // B fragments (32x16 bf16): lane l16 = N; k runs [ks+half*8,+8) and [ks+16+half*8,+8)
            v16bf b_hi[4], b_lo[4];
#pragma unroll
            for (int nt = 0; nt < 4; ++nt) {
                const int ro = (wn * 64 + nt * 16 + l16) * LDSTRH + ks + half * 8;
                const bf8 h0 = *(const bf8*)(ldsWhi + ro);
                const bf8 h1 = *(const bf8*)(ldsWhi + ro + 16);
                const bf8 l0 = *(const bf8*)(ldsWlo + ro);
                const bf8 l1 = *(const bf8*)(ldsWlo + ro + 16);
                b_hi[nt] = CAT16(h0, h1);
                b_lo[nt] = CAT16(l0, l1);
            }

            // Split-bf16 f32 emulation: hi*hi + hi*lo + lo*hi
#pragma unroll
            for (int mt = 0; mt < 2; ++mt)
#pragma unroll
                for (int nt = 0; nt < 4; ++nt) {
                    acc[mt][nt] = __builtin_amdgcn_wmma_f32_16x16x32_bf16(
                        false, a_hi[mt], false, b_hi[nt], (short)0, acc[mt][nt], false, false);
                    acc[mt][nt] = __builtin_amdgcn_wmma_f32_16x16x32_bf16(
                        false, a_hi[mt], false, b_lo[nt], (short)0, acc[mt][nt], false, false);
                    acc[mt][nt] = __builtin_amdgcn_wmma_f32_16x16x32_bf16(
                        false, a_lo[mt], false, b_hi[nt], (short)0, acc[mt][nt], false, false);
                }
        }
        __syncthreads();
    }

    // Epilogue: C/D layout -> N = l16, M = i + 8*half; fused bias + softplus
#pragma unroll
    for (int nt = 0; nt < 4; ++nt) {
        const int gn = bn + wn * 64 + nt * 16 + l16;
        const float bv = bias[gn];
#pragma unroll
        for (int mt = 0; mt < 2; ++mt) {
#pragma unroll
            for (int i = 0; i < 8; ++i) {
                const int gm = bm + wm * 32 + mt * 16 + half * 8 + i;
                const float z = acc[mt][nt][i] + bv;
                const float sp = (z > 20.f) ? z : __logf(1.f + __expf(z));
                __builtin_nontemporal_store(sp, Delta + (size_t)gm * D_MODEL + gn);
            }
        }
    }
}

// ---------------------------------------------------------------------------
// Kernel 3: streaming state update (memory-bound: ~0.58 GB -> ~25 us roofline)
// One lane per (b, d); 16-float contiguous state row via 4x b128 NT accesses.
// ---------------------------------------------------------------------------
__global__ __launch_bounds__(256)
void ssm_update_kernel(const float* __restrict__ x_t,
                       const float* __restrict__ state_prev,
                       const float* __restrict__ Aneg,
                       const float* __restrict__ Bt,
                       const float* __restrict__ Ct,
                       const float* __restrict__ Delta,
                       const float* __restrict__ Dp,
                       float* __restrict__ y_out,
                       float* __restrict__ state_out)
{
    const size_t idx = (size_t)blockIdx.x * 256 + threadIdx.x;   // < BATCH*D_MODEL
    const int    d   = (int)(idx & (D_MODEL - 1));
    const size_t b   = idx >> 11;

    const float delta = __builtin_nontemporal_load(Delta + idx);
    const float xv    = x_t[idx];
    const float dx    = delta * xv;
    float y = Dp[d] * xv;

    const f4* Ar = (const f4*)(Aneg + (size_t)d * D_STATE);
    const f4* Br = (const f4*)(Bt + b * D_STATE);
    const f4* Cr = (const f4*)(Ct + b * D_STATE);
    const f4* Sp = (const f4*)(state_prev + idx * D_STATE);
    f4*       So = (f4*)(state_out + idx * D_STATE);

#pragma unroll
    for (int j = 0; j < 4; ++j) {
        const f4 a  = Ar[j];
        const f4 bb = Br[j];
        const f4 c  = Cr[j];
        const f4 sp = __builtin_nontemporal_load(Sp + j);
        f4 st;
        st.x = __expf(delta * a.x) * sp.x + dx * bb.x;
        st.y = __expf(delta * a.y) * sp.y + dx * bb.y;
        st.z = __expf(delta * a.z) * sp.z + dx * bb.z;
        st.w = __expf(delta * a.w) * sp.w + dx * bb.w;
        y += st.x * c.x + st.y * c.y + st.z * c.z + st.w * c.w;
        __builtin_nontemporal_store(st, So + j);
    }
    __builtin_nontemporal_store(y, y_out + idx);
}

// ---------------------------------------------------------------------------
extern "C" void kernel_launch(void* const* d_in, const int* in_sizes, int n_in,
                              void* d_out, int out_size, void* d_ws, size_t ws_size,
                              hipStream_t stream)
{
    const float* x_t   = (const float*)d_in[0];   // (BATCH, D_MODEL)
    const float* sprev = (const float*)d_in[1];   // (BATCH, D_MODEL, D_STATE)
    const float* A_log = (const float*)d_in[2];   // (D_MODEL, D_STATE)
    const float* W_B   = (const float*)d_in[3];   // (D_STATE, D_MODEL)
    const float* W_C   = (const float*)d_in[4];   // (D_STATE, D_MODEL)
    const float* W_D   = (const float*)d_in[5];   // (D_MODEL, D_MODEL)
    const float* b_D   = (const float*)d_in[6];   // (D_MODEL)
    const float* Dp    = (const float*)d_in[7];   // (D_MODEL)

    float* y_out = (float*)d_out;                                      // BATCH*D_MODEL
    float* s_out = (float*)d_out + (size_t)BATCH * D_MODEL;            // BATCH*D_MODEL*D_STATE

    // Workspace layout: Delta (16 MiB) | Bt | Ct | Aneg  (~16.4 MiB total)
    float* Delta = (float*)d_ws;
    float* BtW   = Delta + (size_t)BATCH * D_MODEL;
    float* CtW   = BtW + (size_t)BATCH * D_STATE;
    float* Aneg  = CtW + (size_t)BATCH * D_STATE;

    ssm_prep_A_kernel<<<(D_MODEL * D_STATE + 255) / 256, 256, 0, stream>>>(
        A_log, Aneg, D_MODEL * D_STATE);

    ssm_bc_kernel<<<BATCH / 8, 256, 0, stream>>>(x_t, W_B, W_C, BtW, CtW);

    dim3 ggrid(BATCH / TILE_M, D_MODEL / TILE_N);   // (16, 16)
    ssm_delta_gemm_softplus_kernel<<<ggrid, 256, 0, stream>>>(x_t, W_D, b_D, Delta);

    ssm_update_kernel<<<(BATCH * D_MODEL) / 256, 256, 0, stream>>>(
        x_t, sprev, Aneg, BtW, CtW, Delta, Dp, y_out, s_out);
}
